// RNN_LSTM_87488483819578
// MI455X (gfx1250) — matrix operand order repaired and checked by reading, not compiled, not run on hardware
//
#include <hip/hip_runtime.h>
#include <hip/hip_bf16.h>

// Problem dims
#define B_   128
#define T_   1024
#define H_   768
#define C_   256
#define G4_  3072          // 4*H
// Partitioning
#define NB_  24            // persistent workgroups (one WGP each)
#define HB_  32            // hidden units per workgroup
#define RW_  128           // 4*HB_ gate rows per workgroup
#define KT_  24            // H_/32 k-tiles (bf16 WMMA K=32)
#define NT_  8             // RW_/16 n-tiles per workgroup
#define TC_  32            // timesteps per projection chunk
#define NCH_ (T_ / TC_)
#define WAVES_ 8

typedef __attribute__((ext_vector_type(16))) __bf16 bf16x16;
typedef __attribute__((ext_vector_type(8)))  float  f32x8;
typedef __attribute__((ext_vector_type(4)))  unsigned int u32x4;
typedef unsigned short u16;
typedef unsigned int   u32;

union Frag { bf16x16 v; u32x4 q[2]; };

__device__ inline f32x8 zero8() {
  f32x8 z = {0.f, 0.f, 0.f, 0.f, 0.f, 0.f, 0.f, 0.f};
  return z;
}
__device__ inline u16 f2bf(float f) {          // round-to-nearest-even f32->bf16
  u32 u = __float_as_uint(f);
  u = u + 0x7FFFu + ((u >> 16) & 1u);
  return (u16)(u >> 16);
}
__device__ inline float sigm(float x) {
  return __builtin_amdgcn_rcpf(1.0f + __expf(-x));
}
__device__ inline float tanh_f(float x) {
  return 1.0f - 2.0f * __builtin_amdgcn_rcpf(__expf(2.0f * x) + 1.0f);
}
__device__ inline f32x8 wmma_bf16(bf16x16 a, bf16x16 b, f32x8 c) {
  // (neg_a, A, neg_b, B, c_mod, C, reuse_a, reuse_b)
  return __builtin_amdgcn_wmma_f32_16x16x32_bf16(false, a, false, b, (short)0, c,
                                                 false, false);
}

// ---------------------------------------------------------------------------
// Setup: bf16-convert W_hh and fc_W, fuse biases, zero h0 + barrier counter.
// ---------------------------------------------------------------------------
__global__ void lstm_setup(const float* __restrict__ W_ih,
                           const float* __restrict__ W_hh,
                           const float* __restrict__ b_ih,
                           const float* __restrict__ b_hh,
                           const float* __restrict__ fc_W,
                           u16* Whh_bf, u16* fcW_bf, float* bias, float* win,
                           u16* hbuf, u32* cnt) {
  int i = blockIdx.x * blockDim.x + threadIdx.x;
  if (i < G4_ * H_) Whh_bf[i] = f2bf(W_hh[i]);
  if (i < C_ * H_)  fcW_bf[i] = f2bf(fc_W[i]);
  if (i < G4_) { bias[i] = b_ih[i] + b_hh[i]; win[i] = W_ih[i]; }
  if (i < 2 * B_ * H_) hbuf[i] = 0;
  if (i == 0) *cnt = 0;
}

// ---------------------------------------------------------------------------
// Fused projection of one finished chunk of cell states: out = cs @ fc_W.T + b
// Software-pipelined: fragments for k-tile kt+1 are loaded before wmma(kt).
// ---------------------------------------------------------------------------
__device__ inline void project_chunk(int ci, const u16* __restrict__ cstage,
                                     const u16* __restrict__ fcW_bf,
                                     const float* __restrict__ fc_b,
                                     float* __restrict__ out,
                                     int g, int wid, int ln, int hi) {
  const u16* cs = cstage + (size_t)(ci & 1) * (TC_ * B_ * H_);
  const int gw = g * WAVES_ + wid;                   // 0..191 global wave id
  const int ntiles = (TC_ * B_ / 16) * (C_ / 16);    // 4096 output tiles
#pragma unroll 1
  for (int tile = gw; tile < ntiles; tile += NB_ * WAVES_) {
    const int mt = tile >> 4;
    const int nt = tile & 15;
    f32x8 acc = zero8();
    const u16* abase = cs + (mt * 16 + ln) * H_ + hi * 8;       // A: K 0-7/16-23 | 8-15/24-31
    const u16* bbase = fcW_bf + (nt * 16 + ln) * H_ + hi * 8;   // B: K 0-7/16-23 | 8-15/24-31
    Frag a, b, an, bn;
    a.q[0] = *(const u32x4*)(abase);
    a.q[1] = *(const u32x4*)(abase + 16);
    b.q[0] = *(const u32x4*)(bbase);
    b.q[1] = *(const u32x4*)(bbase + 16);
#pragma unroll 2
    for (int kt = 0; kt < KT_; ++kt) {
      const int kn = (kt + 1 < KT_) ? (kt + 1) : kt;   // clamped prefetch index
      an.q[0] = *(const u32x4*)(abase + kn * 32);
      an.q[1] = *(const u32x4*)(abase + kn * 32 + 16);
      bn.q[0] = *(const u32x4*)(bbase + kn * 32);
      bn.q[1] = *(const u32x4*)(bbase + kn * 32 + 16);
      acc = wmma_bf16(a.v, b.v, acc);
      a = an;
      b = bn;
    }
    const int bcol = nt * 16 + ln;
    const float fb = fc_b[bcol];
    const int tb = ci * TC_;
#pragma unroll
    for (int r = 0; r < 8; ++r) {
      int row = mt * 16 + r + hi * 8;   // row = t_local*128 + b
      int tl  = row >> 7;
      int bb  = row & 127;
      out[(size_t)(bb * T_ + tb + tl) * C_ + bcol] = acc[r] + fb;
    }
  }
}

// ---------------------------------------------------------------------------
// Persistent LSTM: W_hh slice in LDS, c in VGPRs, grid barrier per timestep.
// ---------------------------------------------------------------------------
__global__ __launch_bounds__(256) void lstm_persistent(
    const float* __restrict__ x, const u16* __restrict__ Whh_bf,
    const u16* __restrict__ fcW_bf, const float* __restrict__ bias,
    const float* __restrict__ win, const float* __restrict__ fc_b,
    u16* hbuf, u16* cstage, float* __restrict__ out, u32* cnt) {

  __shared__ u16 Wl[RW_ * H_];   // 128 x 768 bf16 = 192 KB (LDS-resident weights)

  const int g    = blockIdx.x;
  const int tid  = threadIdx.x;
  const int lane = tid & 31;
  const int wid  = tid >> 5;     // wave id = M-tile (16 batch rows per wave)
  const int ln   = lane & 15;
  const int hi   = lane >> 4;

  // Stage this WG's 128 gate rows {j, H+j, 2H+j, 3H+j} into LDS, row-major.
  {
    const u32* src = (const u32*)Whh_bf;
    u32* dst = (u32*)Wl;
#pragma unroll 1
    for (int i = tid; i < RW_ * H_ / 2; i += 256) {
      int e  = i * 2;
      int lr = e / H_;
      int k  = e - lr * H_;
      int gr = (lr >> 5) * H_ + g * HB_ + (lr & (HB_ - 1));
      dst[i] = src[((size_t)gr * H_ + k) >> 1];
    }
  }
  __syncthreads();

  // Per-lane gate constants (bias and input weight per owned hidden column).
  float bi[2], bff[2], bgg[2], boo[2], wii[2], wff[2], wgg[2], woo[2];
#pragma unroll
  for (int q = 0; q < 2; ++q) {
    int jg = g * HB_ + q * 16 + ln;
    bi[q]  = bias[jg];           wii[q] = win[jg];
    bff[q] = bias[H_ + jg];      wff[q] = win[H_ + jg];
    bgg[q] = bias[2 * H_ + jg];  wgg[q] = win[2 * H_ + jg];
    boo[q] = bias[3 * H_ + jg];  woo[q] = win[3 * H_ + jg];
  }

  float creg[2][8];              // cell state, resident in VGPRs
#pragma unroll
  for (int q = 0; q < 2; ++q)
#pragma unroll
    for (int r = 0; r < 8; ++r) creg[q][r] = 0.0f;

  const int b0 = wid * 16 + ln;

#pragma unroll 1
  for (int t = 0; t < T_; ++t) {
    if (t > 0) {                 // wait until every WG finished step t-1
      if (tid == 0) {
        while (__hip_atomic_load(cnt, __ATOMIC_ACQUIRE, __HIP_MEMORY_SCOPE_AGENT)
               < (u32)(NB_ * t))
          __builtin_amdgcn_s_sleep(1);
      }
      __syncthreads();
      __threadfence();
    }

    // Project the previous (fully written, double-buffered) chunk of c.
    if (t >= TC_ && (t & (TC_ - 1)) == 0)
      project_chunk(t / TC_ - 1, cstage, fcW_bf, fc_b, out, g, wid, ln, hi);

    const u16* hprev = hbuf + (size_t)(t & 1) * (B_ * H_);
    u16* hnext = hbuf + (size_t)((t + 1) & 1) * (B_ * H_);

    // gates[b, rows] = h @ W_hh_slice.T   (8 M=16 tiles across 8 waves)
    f32x8 acc[NT_];
#pragma unroll
    for (int nt = 0; nt < NT_; ++nt) acc[nt] = zero8();

    // A-fragment (h tile) global loads pipelined one k-tile ahead, so the
    // L2 latency is hidden behind the 8 WMMAs of the previous k-tile.
    const u16* abase = hprev + (size_t)b0 * H_ + hi * 8;
    Frag a, an;
    a.q[0] = *(const u32x4*)(abase);
    a.q[1] = *(const u32x4*)(abase + 16);
#pragma unroll 2
    for (int kt = 0; kt < KT_; ++kt) {
      const int kn = (kt + 1 < KT_) ? (kt + 1) : kt;
      an.q[0] = *(const u32x4*)(abase + kn * 32);
      an.q[1] = *(const u32x4*)(abase + kn * 32 + 16);
#pragma unroll
      for (int nt = 0; nt < NT_; ++nt) {
        Frag bb;
        const u16* bp = &Wl[(nt * 16 + ln) * H_ + kt * 32 + hi * 8];
        bb.q[0] = *(const u32x4*)(bp);
        bb.q[1] = *(const u32x4*)(bp + 16);
        acc[nt] = wmma_bf16(a.v, bb.v, acc[nt]);
      }
      a = an;
    }

    float xv[8];
#pragma unroll
    for (int r = 0; r < 8; ++r) {
      int b = wid * 16 + r + hi * 8;
      xv[r] = x[(size_t)b * T_ + t];
    }
    if (t + 1 < T_) __builtin_prefetch(&x[(size_t)b0 * T_ + t + 1], 0, 1);

    u16* cdst = cstage + (size_t)((t / TC_) & 1) * (TC_ * B_ * H_)
                       + (size_t)(t & (TC_ - 1)) * (B_ * H_);

    // i/f/g/o for (b, j) live in the same lane slot of acc[q], acc[q+2],
    // acc[q+4], acc[q+6] -> pure per-lane elementwise LSTM cell update.
#pragma unroll
    for (int q = 0; q < 2; ++q) {
      int jg = g * HB_ + q * 16 + ln;
#pragma unroll
      for (int r = 0; r < 8; ++r) {
        int b = wid * 16 + r + hi * 8;
        float gi = sigm  (acc[q    ][r] + xv[r] * wii[q] + bi[q]);
        float gf = sigm  (acc[q + 2][r] + xv[r] * wff[q] + bff[q]);
        float gg = tanh_f(acc[q + 4][r] + xv[r] * wgg[q] + bgg[q]);
        float go = sigm  (acc[q + 6][r] + xv[r] * woo[q] + boo[q]);
        float c  = gf * creg[q][r] + gi * gg;
        creg[q][r] = c;
        float h  = go * tanh_f(c);
        hnext[(size_t)b * H_ + jg] = f2bf(h);
        cdst [(size_t)b * H_ + jg] = f2bf(c);
      }
    }

    __threadfence();             // make h/c stores agent-visible
    __syncthreads();
    if (tid == 0)
      __hip_atomic_fetch_add(cnt, 1u, __ATOMIC_RELEASE, __HIP_MEMORY_SCOPE_AGENT);
  }

  // Wait for everyone's final step, then project the last chunk.
  if (tid == 0) {
    while (__hip_atomic_load(cnt, __ATOMIC_ACQUIRE, __HIP_MEMORY_SCOPE_AGENT)
           < (u32)(NB_ * T_))
      __builtin_amdgcn_s_sleep(1);
  }
  __syncthreads();
  __threadfence();
  project_chunk(NCH_ - 1, cstage, fcW_bf, fc_b, out, g, wid, ln, hi);
}

// ---------------------------------------------------------------------------
extern "C" void kernel_launch(void* const* d_in, const int* in_sizes, int n_in,
                              void* d_out, int out_size, void* d_ws, size_t ws_size,
                              hipStream_t stream) {
  (void)in_sizes; (void)n_in; (void)out_size; (void)ws_size;
  const float* x    = (const float*)d_in[0];
  const float* W_ih = (const float*)d_in[1];
  const float* W_hh = (const float*)d_in[2];
  const float* b_ih = (const float*)d_in[3];
  const float* b_hh = (const float*)d_in[4];
  const float* fc_W = (const float*)d_in[5];
  const float* fc_b = (const float*)d_in[6];
  float* out = (float*)d_out;

  // Workspace layout (~18.1 MB, all regions 256B-multiple sized)
  char* ws = (char*)d_ws;
  size_t off = 0;
  u16*   Whh_bf = (u16*)  (ws + off); off += (size_t)G4_ * H_ * 2;       // 4.72 MB
  u16*   fcW_bf = (u16*)  (ws + off); off += (size_t)C_  * H_ * 2;       // 0.39 MB
  float* bias   = (float*)(ws + off); off += (size_t)G4_ * 4;
  float* win    = (float*)(ws + off); off += (size_t)G4_ * 4;
  u16*   hbuf   = (u16*)  (ws + off); off += (size_t)2 * B_ * H_ * 2;    // 0.39 MB
  u16*   cstage = (u16*)  (ws + off); off += (size_t)2 * TC_ * B_ * H_ * 2; // 12.6 MB
  u32*   cnt    = (u32*)  (ws + off); off += 256;

  const int n = G4_ * H_;
  lstm_setup<<<(n + 255) / 256, 256, 0, stream>>>(W_ih, W_hh, b_ih, b_hh, fc_W,
                                                  Whh_bf, fcW_bf, bias, win,
                                                  hbuf, cnt);
  lstm_persistent<<<NB_, 256, 0, stream>>>(x, Whh_bf, fcW_bf, bias, win, fc_b,
                                           hbuf, cstage, out, cnt);
}
